// CartNet_49263274885682
// MI455X (gfx1250) — matrix operand order repaired and checked by reading, not compiled, not run on hardware
//
#include <hip/hip_runtime.h>
#include <hip/hip_bf16.h>
#include <stdint.h>

#define N_NODES  12500
#define N_EDGES  200000
#define DIM      128
#define N_GRAPHS 64
#define BN_EPS   1e-5f

typedef __attribute__((ext_vector_type(16))) _Float16 v16h;
typedef __attribute__((ext_vector_type(8)))  float    v8f;

#define DEV static __device__ __forceinline__

DEV float siluf(float x)   { return x / (1.0f + __expf(-x)); }
DEV float sigmf(float x)   { return 1.0f / (1.0f + __expf(-x)); }
DEV float cutoff5(float d) { return (d < 5.0f) ? 0.5f * (__cosf(d * 0.62831853071f) + 1.0f) : 0.0f; }

// ---------------------------------------------------------------------------
// Core WMMA tile: one wave computes a 16x16 f32 tile of  A[16,K] x B[K,16].
// A: LDS f16 row-major (strideA halves, 16B-aligned rows).
//   lane<16 : row=lane,    K = {0..7, 16..23} + 32*s
//   lane>=16: row=lane-15? no: row=lane&15,  K = {8..15, 24..31} + 32*s
// B: pre-swizzled global blob, ((tile*ksteps + s)*32 + lane) * 32 bytes.
// ---------------------------------------------------------------------------
DEV v8f gemm_tile(const _Float16* __restrict__ Abase, int strideA,
                  const uint32_t* __restrict__ Bpack, int tile, int ksteps, int lane)
{
    v8f c = {};
    const int row  = lane & 15;
    const int koff = (lane >> 4) << 3;               // 0 or 8
    const _Float16* arow = Abase + row * strideA + koff;
    const v16h* bp = (const v16h*)Bpack + (size_t)tile * ksteps * 32 + lane;
    for (int s = 0; s < ksteps; ++s) {
        v16h a;
#pragma unroll
        for (int i = 0; i < 8; ++i) a[i]     = arow[i];
#pragma unroll
        for (int i = 0; i < 8; ++i) a[8 + i] = arow[16 + i];
        v16h b = bp[(size_t)s * 32];
        c = __builtin_amdgcn_wmma_f32_16x16x32_f16(false, a, false, b, (short)0, c,
                                                   false, false);
        arow += 32;
    }
    return c;
}

// ---------------------------------------------------------------------------
// Weight pre-pack: fp32 W[K][Nn] (row-major, fin x fout) -> B-operand blob,
// zero-padded to Kpad (multiple of 32). One thread per packed u32.
// out[((t*KS + s)*32 + lane)*8 + i] packs f16(W[k0][n]) | f16(W[k0+1][n])<<16
// with n = t*16 + (lane&15), k0 = s*32 + ((lane>=16)?8:0) + 2*i + ((i>=4)?8:0).
// ---------------------------------------------------------------------------
__global__ void k_prepack(const float* __restrict__ W, uint32_t* __restrict__ out,
                          int K, int Kpad, int Nn)
{
    int total = (Nn >> 4) * (Kpad >> 5) * 32 * 8;
    int idx = blockIdx.x * blockDim.x + threadIdx.x;
    if (idx >= total) return;
    int i    = idx & 7;
    int lane = (idx >> 3) & 31;
    int rest = idx >> 8;
    int KS   = Kpad >> 5;
    int s    = rest % KS;
    int t    = rest / KS;
    int n    = t * 16 + (lane & 15);
    int k0   = s * 32 + ((lane >> 4) << 3) + 2 * i + ((i >> 2) << 3);
    float w0 = (k0     < K) ? W[(size_t)k0       * Nn + n] : 0.0f;
    float w1 = (k0 + 1 < K) ? W[(size_t)(k0 + 1) * Nn + n] : 0.0f;
    union { _Float16 h[2]; uint32_t u; } p;
    p.h[0] = (_Float16)w0; p.h[1] = (_Float16)w1;
    out[idx] = p.u;
}

// ---------------------------------------------------------------------------
// Node encoder: x = silu(silu(emb[z] + t[batch]) @ encW + enc_b)   [N,128]
// 16 nodes per block, 8 waves each own one 16-col tile; K=256 (8 ksteps).
// ---------------------------------------------------------------------------
__global__ void __launch_bounds__(256)
k_node_encoder(const int* __restrict__ atom_z, const int* __restrict__ batch_idx,
               const float* __restrict__ temperature, const float* __restrict__ emb,
               const float* __restrict__ temp_W, const float* __restrict__ temp_b,
               const uint32_t* __restrict__ encP, const float* __restrict__ enc_b,
               float* __restrict__ x_out)
{
    __shared__ _Float16 A[16][264];                 // 256 + 8 pad (16B-aligned rows)
    int nb = blockIdx.x * 16;
    for (int e = threadIdx.x; e < 16 * 256; e += 256) {
        int r = e >> 8, cc = e & 255;
        int node = nb + r;
        float v = 0.0f;
        if (node < N_NODES) {
            int z = atom_z[node], g = batch_idx[node];
            v = siluf(emb[(size_t)z * 256 + cc] + temperature[g] * temp_W[cc] + temp_b[cc]);
        }
        A[r][cc] = (_Float16)v;
    }
    __syncthreads();
    int wave = threadIdx.x >> 5, lane = threadIdx.x & 31;
    v8f c = gemm_tile(&A[0][0], 264, encP, wave, 8, lane);
    int n = wave * 16 + (lane & 15);
    int rb = (lane >> 4) << 3;
    float b = enc_b[n];
#pragma unroll
    for (int r = 0; r < 8; ++r) {
        int node = nb + rb + r;
        if (node < N_NODES) x_out[(size_t)node * DIM + n] = siluf(c[r] + b);
    }
}

// ---------------------------------------------------------------------------
// Edge encoder: feat = [rbf(d)(64) | dir(3) | 0-pad] (K=96),
// e = silu(silu(feat @ W1 + b1) @ W2 + b2)                         [E,128]
// ---------------------------------------------------------------------------
__global__ void __launch_bounds__(256)
k_edge_encoder(const float* __restrict__ cart_dist, const float* __restrict__ cart_dir,
               const uint32_t* __restrict__ W1P, const float* __restrict__ b1,
               const uint32_t* __restrict__ W2P, const float* __restrict__ b2,
               float* __restrict__ e_out)
{
    __shared__ _Float16 A[16][104];                 // 96 + 8 pad
    __shared__ _Float16 M[16][264];                 // 256 + 8 pad
    int eb = blockIdx.x * 16;
    const float startc = 0.006737947f;              // exp(-5)
    const float mstep  = (1.0f - startc) * (1.0f / 63.0f);
    const float tbw    = (2.0f / 64.0f) * (1.0f - startc);
    const float beta   = 1.0f / (tbw * tbw);
    for (int idx = threadIdx.x; idx < 16 * 96; idx += 256) {
        int r = idx / 96, cc = idx % 96;
        int e = eb + r;
        float d = cart_dist[e];
        float v = 0.0f;
        if (cc < 64) {
            float u = __expf(-d) - (startc + cc * mstep);
            v = cutoff5(d) * __expf(-beta * u * u);
        } else if (cc < 67) {
            v = cart_dir[(size_t)e * 3 + (cc - 64)];
        }
        A[r][cc] = (_Float16)v;
    }
    __syncthreads();
    int wave = threadIdx.x >> 5, lane = threadIdx.x & 31;
    int col = lane & 15, rb = (lane >> 4) << 3;
#pragma unroll
    for (int tt = 0; tt < 2; ++tt) {                // 16 output tiles (N=256), 2 per wave
        int t = wave + tt * 8;
        v8f c1 = gemm_tile(&A[0][0], 104, W1P, t, 3, lane);
        int n = t * 16 + col;
        float bb = b1[n];
#pragma unroll
        for (int r = 0; r < 8; ++r) M[rb + r][n] = (_Float16)siluf(c1[r] + bb);
    }
    __syncthreads();
    v8f c2 = gemm_tile(&M[0][0], 264, W2P, wave, 8, lane);
    int n = wave * 16 + col;
    float bb = b2[n];
#pragma unroll
    for (int r = 0; r < 8; ++r)
        e_out[(size_t)(eb + rb + r) * DIM + n] = siluf(c2[r] + bb);
}

// ---------------------------------------------------------------------------
// Layer MLPs: H = [x[dst] | x[src] | e]  (K=384), shared staging, then
//   gate_pre = silu(H@gW1+gb1)@gW2+gb2 ; sender = silu(H@aW1+ab1)@aW2+ab2
// ---------------------------------------------------------------------------
__global__ void __launch_bounds__(256)
k_layer_mlp(const float* __restrict__ x, const float* __restrict__ efeat,
            const int* __restrict__ edge_index,
            const uint32_t* __restrict__ gW1P, const float* __restrict__ gb1,
            const uint32_t* __restrict__ gW2P, const float* __restrict__ gb2,
            const uint32_t* __restrict__ aW1P, const float* __restrict__ ab1,
            const uint32_t* __restrict__ aW2P, const float* __restrict__ ab2,
            float* __restrict__ gate_pre, float* __restrict__ sender)
{
    __shared__ _Float16 H[16][392];                 // 384 + 8 pad
    __shared__ _Float16 M[16][136];                 // 128 + 8 pad
    int eb = blockIdx.x * 16;
    for (int idx = threadIdx.x; idx < 16 * 384; idx += 256) {
        int r = idx / 384, cc = idx % 384;
        int e = eb + r;
        int src = edge_index[e];
        int dst = edge_index[N_EDGES + e];
        float v;
        if (cc < 128)      v = x[(size_t)dst * DIM + cc];
        else if (cc < 256) v = x[(size_t)src * DIM + (cc - 128)];
        else               v = efeat[(size_t)e * DIM + (cc - 256)];
        H[r][cc] = (_Float16)v;
    }
    __syncthreads();
    int wave = threadIdx.x >> 5, lane = threadIdx.x & 31;
    int col = lane & 15, rb = (lane >> 4) << 3;
    int n = wave * 16 + col;
#pragma unroll
    for (int j = 0; j < 2; ++j) {
        const uint32_t* W1P = j ? aW1P : gW1P;
        const uint32_t* W2P = j ? aW2P : gW2P;
        const float* B1 = j ? ab1 : gb1;
        const float* B2 = j ? ab2 : gb2;
        float* out = j ? sender : gate_pre;
        v8f c1 = gemm_tile(&H[0][0], 392, W1P, wave, 12, lane);
        __syncthreads();                            // M free from previous pass
        float bb1 = B1[n];
#pragma unroll
        for (int r = 0; r < 8; ++r) M[rb + r][n] = (_Float16)siluf(c1[r] + bb1);
        __syncthreads();
        v8f c2 = gemm_tile(&M[0][0], 136, W2P, wave, 4, lane);
        float bb2 = B2[n];
#pragma unroll
        for (int r = 0; r < 8; ++r)
            out[(size_t)(eb + rb + r) * DIM + n] = c2[r] + bb2;
    }
}

// ---------------------------------------------------------------------------
// Utility / reduction kernels
// ---------------------------------------------------------------------------
__global__ void k_zero(float* __restrict__ p, int n)
{
    int i = blockIdx.x * blockDim.x + threadIdx.x;
    if (i < n) p[i] = 0.0f;
}

#define BN_CHUNK 512
__global__ void __launch_bounds__(128)
k_bn_reduce(const float* __restrict__ data, int rows,
            float* __restrict__ s1, float* __restrict__ s2)
{
    int f = threadIdx.x;
    int r0 = blockIdx.x * BN_CHUNK;
    int r1 = r0 + BN_CHUNK; if (r1 > rows) r1 = rows;
    float a = 0.0f, b = 0.0f;
    for (int r = r0; r < r1; ++r) {
        float v = data[(size_t)r * DIM + f];
        a += v; b += v * v;
    }
    atomicAdd(&s1[f], a);
    atomicAdd(&s2[f], b);
}

__global__ void k_bn_finalize(const float* __restrict__ s1, const float* __restrict__ s2,
                              const float* __restrict__ g, const float* __restrict__ b,
                              float inv_rows, float* __restrict__ scale,
                              float* __restrict__ shift)
{
    int f = threadIdx.x;
    float m  = s1[f] * inv_rows;
    float var = s2[f] * inv_rows - m * m;
    float sc = g[f] * rsqrtf(var + BN_EPS);
    scale[f] = sc;
    shift[f] = b[f] - m * sc;
}

__global__ void k_edge_aggregate(const float* __restrict__ gate_pre,
                                 const float* __restrict__ sender,
                                 const float* __restrict__ cart_dist,
                                 const int* __restrict__ edge_index,
                                 const float* __restrict__ scale,
                                 const float* __restrict__ shift,
                                 float* __restrict__ efeat, float* __restrict__ aggr)
{
    size_t idx = (size_t)blockIdx.x * blockDim.x + threadIdx.x;
    if (idx >= (size_t)N_EDGES * DIM) return;
    int e = (int)(idx >> 7);
    int f = (int)(idx & 127);
    float env = cutoff5(cart_dist[e]);
    float sigma = env * sigmf(gate_pre[idx] * scale[f] + shift[f]);
    int dst = edge_index[N_EDGES + e];
    atomicAdd(&aggr[(size_t)dst * DIM + f], sigma * sender[idx]);
    efeat[idx] += sigma;
}

__global__ void k_node_update(const float* __restrict__ aggr,
                              const float* __restrict__ scale,
                              const float* __restrict__ shift, float* __restrict__ x)
{
    size_t idx = (size_t)blockIdx.x * blockDim.x + threadIdx.x;
    if (idx >= (size_t)N_NODES * DIM) return;
    int f = (int)(idx & 127);
    x[idx] = siluf(aggr[idx] * scale[f] + shift[f]) + x[idx];
}

__global__ void __launch_bounds__(64)
k_head(const float* __restrict__ x, const int* __restrict__ batch_idx,
       const float* __restrict__ W1, const float* __restrict__ b1,
       const float* __restrict__ W2, const float* __restrict__ b2,
       float* __restrict__ gsums)
{
    __shared__ float xs[128];
    __shared__ float red[64];
    int node = blockIdx.x;
    int j = threadIdx.x;
    for (int cc = j; cc < 128; cc += 64) xs[cc] = x[(size_t)node * DIM + cc];
    __syncthreads();
    float acc = b1[j];
    for (int k = 0; k < 128; ++k) acc += xs[k] * W1[k * 64 + j];
    red[j] = siluf(acc) * W2[j];
    __syncthreads();
    for (int s = 32; s > 0; s >>= 1) {
        if (j < s) red[j] += red[j + s];
        __syncthreads();
    }
    if (j == 0) atomicAdd(&gsums[batch_idx[node]], red[0] + b2[0]);
}

__global__ void k_count(const int* __restrict__ batch_idx, float* __restrict__ gcnt)
{
    int n = blockIdx.x * blockDim.x + threadIdx.x;
    if (n < N_NODES) atomicAdd(&gcnt[batch_idx[n]], 1.0f);
}

__global__ void k_finalize(const float* __restrict__ gsums, const float* __restrict__ gcnt,
                           float* __restrict__ out)
{
    int g = threadIdx.x;
    out[g] = gsums[g] / fmaxf(gcnt[g], 1.0f);
}

// ---------------------------------------------------------------------------
// Host driver
// ---------------------------------------------------------------------------
extern "C" void kernel_launch(void* const* d_in, const int* in_sizes, int n_in,
                              void* d_out, int out_size, void* d_ws, size_t ws_size,
                              hipStream_t stream)
{
    (void)in_sizes; (void)n_in; (void)out_size; (void)ws_size;
    const int*   atom_z     = (const int*)  d_in[0];
    const int*   batch_idx  = (const int*)  d_in[1];
    const int*   edge_index = (const int*)  d_in[2];
    const float* cart_dist  = (const float*)d_in[3];
    const float* cart_dir   = (const float*)d_in[4];
    const float* temperature= (const float*)d_in[5];
    const float* emb        = (const float*)d_in[6];
    const float* temp_W     = (const float*)d_in[7];
    const float* temp_b     = (const float*)d_in[8];
    const float* enc_atom_W = (const float*)d_in[9];
    const float* enc_atom_b = (const float*)d_in[10];
    const float* edge_W1    = (const float*)d_in[11];
    const float* edge_b1    = (const float*)d_in[12];
    const float* edge_W2    = (const float*)d_in[13];
    const float* edge_b2    = (const float*)d_in[14];
    const float* gate_W1    = (const float*)d_in[15];
    const float* gate_b1    = (const float*)d_in[16];
    const float* gate_W2    = (const float*)d_in[17];
    const float* gate_b2    = (const float*)d_in[18];
    const float* aggr_W1    = (const float*)d_in[19];
    const float* aggr_b1    = (const float*)d_in[20];
    const float* aggr_W2    = (const float*)d_in[21];
    const float* aggr_b2    = (const float*)d_in[22];
    const float* norm1_g    = (const float*)d_in[23];
    const float* norm1_b    = (const float*)d_in[24];
    const float* norm2_g    = (const float*)d_in[25];
    const float* norm2_b    = (const float*)d_in[26];
    const float* head_W1    = (const float*)d_in[27];
    const float* head_b1    = (const float*)d_in[28];
    const float* head_W2    = (const float*)d_in[29];
    const float* head_b2    = (const float*)d_in[30];
    float* out = (float*)d_out;

    // ---- workspace layout (f32 element offsets, 64-elt aligned) ----
    float* base = (float*)d_ws;
    size_t off = 0;
    auto alloc = [&](size_t n) { size_t c = off; off += (n + 63) & ~(size_t)63; return c; };
    float* x       = base + alloc((size_t)N_NODES * DIM);
    float* efeat   = base + alloc((size_t)N_EDGES * DIM);
    float* gatep   = base + alloc((size_t)N_EDGES * DIM);
    float* sender  = base + alloc((size_t)N_EDGES * DIM);
    float* aggr    = base + alloc((size_t)N_NODES * DIM);
    float* s1      = base + alloc(128);
    float* s2      = base + alloc(128);
    float* scl     = base + alloc(128);
    float* shf     = base + alloc(128);
    float* gsums   = base + alloc(64);
    float* gcnt    = base + alloc(64);
    uint32_t* encP   = (uint32_t*)(base + alloc(16384));   // 8t * 8s * 256
    uint32_t* eW1P   = (uint32_t*)(base + alloc(12288));   // 16t * 3s * 256
    uint32_t* eW2P   = (uint32_t*)(base + alloc(16384));
    uint32_t* gW1P[3], *gW2P[3], *aW1P[3], *aW2P[3];
    for (int l = 0; l < 3; ++l) {
        gW1P[l] = (uint32_t*)(base + alloc(24576));        // 8t * 12s * 256
        gW2P[l] = (uint32_t*)(base + alloc(8192));         // 8t * 4s * 256
        aW1P[l] = (uint32_t*)(base + alloc(24576));
        aW2P[l] = (uint32_t*)(base + alloc(8192));
    }

    // ---- weight pre-pack (cheap, re-run every launch for determinism) ----
    auto pack = [&](const float* W, uint32_t* P, int K, int Kpad, int Nn) {
        int total = (Nn >> 4) * (Kpad >> 5) * 256;
        k_prepack<<<(total + 255) / 256, 256, 0, stream>>>(W, P, K, Kpad, Nn);
    };
    pack(enc_atom_W, encP, 256, 256, 128);
    pack(edge_W1,    eW1P,  67,  96, 256);
    pack(edge_W2,    eW2P, 256, 256, 128);
    for (int l = 0; l < 3; ++l) {
        pack(gate_W1 + (size_t)l * 384 * 128, gW1P[l], 384, 384, 128);
        pack(gate_W2 + (size_t)l * 128 * 128, gW2P[l], 128, 128, 128);
        pack(aggr_W1 + (size_t)l * 384 * 128, aW1P[l], 384, 384, 128);
        pack(aggr_W2 + (size_t)l * 128 * 128, aW2P[l], 128, 128, 128);
    }

    // ---- encoders ----
    k_node_encoder<<<(N_NODES + 15) / 16, 256, 0, stream>>>(
        atom_z, batch_idx, temperature, emb, temp_W, temp_b, encP, enc_atom_b, x);
    k_edge_encoder<<<N_EDGES / 16, 256, 0, stream>>>(
        cart_dist, cart_dir, eW1P, edge_b1, eW2P, edge_b2, efeat);

    // ---- CartNet layers ----
    for (int l = 0; l < 3; ++l) {
        k_layer_mlp<<<N_EDGES / 16, 256, 0, stream>>>(
            x, efeat, edge_index,
            gW1P[l], gate_b1 + l * 128, gW2P[l], gate_b2 + l * 128,
            aW1P[l], aggr_b1 + l * 128, aW2P[l], aggr_b2 + l * 128,
            gatep, sender);

        // edge batchnorm on gate
        k_zero<<<1, 256, 0, stream>>>(s1, 256);            // s1,s2 contiguous
        k_bn_reduce<<<(N_EDGES + BN_CHUNK - 1) / BN_CHUNK, 128, 0, stream>>>(
            gatep, N_EDGES, s1, s2);
        k_bn_finalize<<<1, 128, 0, stream>>>(s1, s2, norm1_g + l * 128, norm1_b + l * 128,
                                             1.0f / (float)N_EDGES, scl, shf);

        // message aggregation (segment_sum by dst) + e update
        k_zero<<<((size_t)N_NODES * DIM + 255) / 256, 256, 0, stream>>>(aggr, N_NODES * DIM);
        k_edge_aggregate<<<((size_t)N_EDGES * DIM + 255) / 256, 256, 0, stream>>>(
            gatep, sender, cart_dist, edge_index, scl, shf, efeat, aggr);

        // node batchnorm + residual
        k_zero<<<1, 256, 0, stream>>>(s1, 256);
        k_bn_reduce<<<(N_NODES + BN_CHUNK - 1) / BN_CHUNK, 128, 0, stream>>>(
            aggr, N_NODES, s1, s2);
        k_bn_finalize<<<1, 128, 0, stream>>>(s1, s2, norm2_g + l * 128, norm2_b + l * 128,
                                             1.0f / (float)N_NODES, scl, shf);
        k_node_update<<<((size_t)N_NODES * DIM + 255) / 256, 256, 0, stream>>>(
            aggr, scl, shf, x);
    }

    // ---- scalar head + per-graph mean ----
    k_zero<<<1, 128, 0, stream>>>(gsums, 128);             // gsums,gcnt contiguous
    k_count<<<(N_NODES + 255) / 256, 256, 0, stream>>>(batch_idx, gcnt);
    k_head<<<N_NODES, 64, 0, stream>>>(x, batch_idx, head_W1, head_b1, head_W2, head_b2,
                                       gsums);
    k_finalize<<<1, 64, 0, stream>>>(gsums, gcnt, out);
}
